// model94_66881230733566
// MI455X (gfx1250) — matrix operand order repaired and checked by reading, not compiled, not run on hardware
//
#include <hip/hip_runtime.h>

typedef _Float16 half_t;
typedef __attribute__((ext_vector_type(16))) _Float16 v16h;
typedef __attribute__((ext_vector_type(8)))  _Float16 v8h;
typedef __attribute__((ext_vector_type(8)))  float    v8f;

#define NN     94     // nodes per graph
#define KPAD   96     // K padded to 3 x 32 for wmma 16x16x32
#define NOUT   6400   // fc output width
#define EPMAX  1600   // max edges incl. self loops (1504 + 94 = 1598)

// ---------------------------------------------------------------------------
// Kernel 1: graph preprocessing -> symmetric norm + CSR sorted by destination.
// Serial sections run on thread 0 for bitwise determinism (tiny: ~3k iters).
// ---------------------------------------------------------------------------
__global__ void prep_graph_kernel(const int* __restrict__ edge_index,
                                  const float* __restrict__ edge_weight,
                                  int E,
                                  int* __restrict__ g_rowptr,
                                  int* __restrict__ g_col,
                                  float* __restrict__ g_val) {
  __shared__ int   sSrc[EPMAX];
  __shared__ int   sDst[EPMAX];
  __shared__ float sEw [EPMAX];
  __shared__ float sNorm[EPMAX];
  __shared__ float sDeg[NN];
  __shared__ float sDinv[NN];
  __shared__ int   sCnt[NN];
  __shared__ int   sPtr[NN + 1];
  const int tid = threadIdx.x;
  int EP = E + NN;
  if (EP > EPMAX) EP = EPMAX;

  for (int e = tid; e < EP; e += blockDim.x) {
    int s, d; float w;
    if (e < E) { s = edge_index[e]; d = edge_index[E + e]; w = edge_weight[e]; }
    else       { s = e - E;         d = e - E;             w = 1.0f;           }
    sSrc[e] = s; sDst[e] = d; sEw[e] = w;
  }
  for (int n = tid; n < NN; n += blockDim.x) { sDeg[n] = 0.0f; sCnt[n] = 0; }
  __syncthreads();

  if (tid == 0) {  // deterministic degree accumulation
    for (int e = 0; e < EP; ++e) sDeg[sDst[e]] += sEw[e];
  }
  __syncthreads();

  for (int n = tid; n < NN; n += blockDim.x) {
    float d = sDeg[n];
    sDinv[n] = (d > 0.0f) ? rsqrtf(d) : 0.0f;
  }
  __syncthreads();

  for (int e = tid; e < EP; e += blockDim.x)
    sNorm[e] = sDinv[sSrc[e]] * sEw[e] * sDinv[sDst[e]];
  __syncthreads();

  if (tid == 0) {  // counting sort by destination (stable, serial, deterministic)
    for (int e = 0; e < EP; ++e) sCnt[sDst[e]]++;
    int run = 0;
    for (int n = 0; n < NN; ++n) { sPtr[n] = run; run += sCnt[n]; }
    sPtr[NN] = run;
    for (int n = 0; n < NN; ++n) sCnt[n] = sPtr[n];  // running cursors
    for (int e = 0; e < EP; ++e) {
      int d = sDst[e];
      int p = sCnt[d]++;
      g_col[p] = sSrc[e];
      g_val[p] = sNorm[e];
    }
    for (int n = 0; n <= NN; ++n) g_rowptr[n] = sPtr[n];
  }
}

// ---------------------------------------------------------------------------
// Kernel 2: repack Wfc[6400,94] (row-major [out,in]) into fp16 WMMA-B order:
// Bpack[kt][col][kk], kt in 0..2, kk = k within 32-wide k-tile, zero padded.
// A lane's B fragment is then one contiguous 32-byte read.
// ---------------------------------------------------------------------------
__global__ void pack_wfc_kernel(const float* __restrict__ Wfc,
                                half_t* __restrict__ Bpack) {
  int idx = blockIdx.x * blockDim.x + threadIdx.x;
  const int total = 3 * NOUT * 32;
  if (idx >= total) return;
  int kt  = idx / (NOUT * 32);
  int r   = idx % (NOUT * 32);
  int col = r >> 5;
  int kk  = r & 31;
  int k   = kt * 32 + kk;
  float v = (k < NN) ? Wfc[col * NN + k] : 0.0f;
  Bpack[idx] = (half_t)v;
}

// ---------------------------------------------------------------------------
// Kernel 3: fused GCN conv1(tanh) + conv2(tanh) per batch element.
// CSR cached in LDS; output is fp16 A rows of length KPAD (pad cols = 0).
// ---------------------------------------------------------------------------
__global__ void __launch_bounds__(128) gcn_conv_kernel(
    const float* __restrict__ feature,
    const float* __restrict__ W1, const float* __restrict__ b1,
    const float* __restrict__ W2, const float* __restrict__ b2,
    const int* __restrict__ rowptr, const int* __restrict__ colsrc,
    const float* __restrict__ val, int EP,
    half_t* __restrict__ Apack) {
  __shared__ float sF [NN * 3];
  __shared__ float sXW[NN * 6];
  __shared__ float sH1[NN * 6];
  __shared__ float sHW[NN];
  __shared__ float sVal[EPMAX];
  __shared__ int   sCol[EPMAX];
  __shared__ int   sRow[NN + 1];
  __shared__ float sW1[18], sB1[6], sW2[6], sB2;

  const int tid = threadIdx.x;
  const int b   = blockIdx.x;
  const float* f = feature + (size_t)b * (NN * 3);

  for (int i = tid; i < NN * 3; i += blockDim.x) sF[i] = f[i];
  for (int i = tid; i < EP; i += blockDim.x) { sVal[i] = val[i]; sCol[i] = colsrc[i]; }
  for (int i = tid; i <= NN; i += blockDim.x) sRow[i] = rowptr[i];
  if (tid < 18) sW1[tid] = W1[tid];
  if (tid < 6)  { sB1[tid] = b1[tid]; sW2[tid] = W2[tid]; }
  if (tid == 0) sB2 = b2[0];
  __syncthreads();

  // x @ W1  -> sXW[94,6]
  for (int idx = tid; idx < NN * 6; idx += blockDim.x) {
    int n = idx / 6, c = idx % 6;
    sXW[idx] = sF[n * 3 + 0] * sW1[0 * 6 + c] +
               sF[n * 3 + 1] * sW1[1 * 6 + c] +
               sF[n * 3 + 2] * sW1[2 * 6 + c];
  }
  __syncthreads();

  // aggregate + bias + tanh -> sH1[94,6]
  for (int idx = tid; idx < NN * 6; idx += blockDim.x) {
    int n = idx / 6, c = idx % 6;
    float s = 0.0f;
    int e0 = sRow[n], e1 = sRow[n + 1];
    for (int e = e0; e < e1; ++e) s += sVal[e] * sXW[sCol[e] * 6 + c];
    sH1[idx] = tanhf(s + sB1[c]);
  }
  __syncthreads();

  // h1 @ W2 -> sHW[94]
  for (int n = tid; n < NN; n += blockDim.x) {
    float s = 0.0f;
    #pragma unroll
    for (int c = 0; c < 6; ++c) s += sH1[n * 6 + c] * sW2[c];
    sHW[n] = s;
  }
  __syncthreads();

  // aggregate + bias + tanh -> fp16 A row (padded to KPAD)
  for (int n = tid; n < KPAD; n += blockDim.x) {
    float h2 = 0.0f;
    if (n < NN) {
      float s = 0.0f;
      int e0 = sRow[n], e1 = sRow[n + 1];
      for (int e = e0; e < e1; ++e) s += sVal[e] * sHW[sCol[e]];
      h2 = tanhf(s + sB2);
    }
    Apack[(size_t)b * KPAD + n] = (half_t)h2;
  }
}

// ---------------------------------------------------------------------------
// Kernel 4: C[M,6400] = A[M,94] x Wfc^T + bfc via v_wmma_f32_16x16x32_f16.
// 4 waves/block; each wave: one 16-row M-tile x four 16-col N-tiles,
// A fragment reused across the 4 N-tiles; K = 3 tiles of 32 (zero padded).
// Output stores are non-temporal: the 419 MB fp32 stream must not evict the
// L2-resident Apack/Bpack operands (L2 = 192 MB).
// ---------------------------------------------------------------------------
__global__ void __launch_bounds__(128) gemm_wmma_kernel(
    const half_t* __restrict__ Apack, const half_t* __restrict__ Bpack,
    const float* __restrict__ bfc, float* __restrict__ out, int M) {
  const int lane  = threadIdx.x & 31;
  const int wave  = threadIdx.x >> 5;
  const int mtile = blockIdx.y * 4 + wave;      // wave-uniform
  if (mtile * 16 >= M) return;                  // uniform guard: EXEC stays all-1s
  const int ntile0 = blockIdx.x * 4;
  const int hh   = lane >> 4;                   // half-wave select
  const int mrow = mtile * 16 + (lane & 15);

  v8f acc[4] = {v8f{}, v8f{}, v8f{}, v8f{}};

  #pragma unroll
  for (int kt = 0; kt < 3; ++kt) {
    // A fragment in this code's contiguous k order: lane holds row (lane&15);
    // elements 0..15 cover K = 16*hh + 0..15 (matches B's contiguous order)
    // -> two adjacent 16B chunks of the row.
    const half_t* arow = Apack + (size_t)mrow * KPAD + kt * 32;
    v8h c0 = *(const v8h*)(arow + 16 * hh);
    v8h c1 = *(const v8h*)(arow + 16 * hh + 8);
    v16h a;
    #pragma unroll
    for (int i = 0; i < 8; ++i) { a[i] = c0[i]; a[8 + i] = c1[i]; }

    #pragma unroll
    for (int j = 0; j < 4; ++j) {
      // B fragment: lane holds column (lane&15); halves 0..15 = K 16*hh..+15,
      // contiguous in our packed layout -> one 32B read.
      int col = (ntile0 + j) * 16 + (lane & 15);
      const half_t* bp = Bpack + ((size_t)kt * NOUT + col) * 32 + 16 * hh;
      v16h bb = *(const v16h*)bp;
      acc[j] = __builtin_amdgcn_wmma_f32_16x16x32_f16(
          /*neg_a=*/false, a, /*neg_b=*/false, bb,
          /*c_mod=*/(short)0, acc[j], /*reuse_a=*/false, /*reuse_b=*/false);
    }
  }

  // C/D layout: VGPR r -> M = r + 8*hh, N = lane&15. Fuse bias add.
  // Non-temporal stores: streaming output, never re-read by this kernel.
  #pragma unroll
  for (int j = 0; j < 4; ++j) {
    int col = (ntile0 + j) * 16 + (lane & 15);
    float bias = bfc[col];
    #pragma unroll
    for (int r = 0; r < 8; ++r) {
      int m = mtile * 16 + r + 8 * hh;
      __builtin_nontemporal_store(acc[j][r] + bias,
                                  &out[(size_t)m * NOUT + col]);
    }
  }
}

// ---------------------------------------------------------------------------
extern "C" void kernel_launch(void* const* d_in, const int* in_sizes, int n_in,
                              void* d_out, int out_size, void* d_ws, size_t ws_size,
                              hipStream_t stream) {
  (void)n_in; (void)out_size; (void)ws_size;
  const float* feature     = (const float*)d_in[0];
  const int*   edge_index  = (const int*)  d_in[1];
  const float* edge_weight = (const float*)d_in[2];
  const float* W1  = (const float*)d_in[3];
  const float* b1  = (const float*)d_in[4];
  const float* W2  = (const float*)d_in[5];
  const float* b2  = (const float*)d_in[6];
  const float* Wfc = (const float*)d_in[7];
  const float* bfc = (const float*)d_in[8];
  float* out = (float*)d_out;

  const int E    = in_sizes[2];                // edge count (1504)
  int EP = E + NN; if (EP > EPMAX) EP = EPMAX; // with self loops (1598)
  const int Btot = in_sizes[0] / (NN * 3);     // 16384

  // Workspace layout (256B-aligned regions)
  char* ws = (char*)d_ws;
  int*    rowptr = (int*)(ws + 0);                         // 95 ints
  int*    colsrc = (int*)(ws + 512);                       // <=1600 ints
  float*  val    = (float*)(ws + 7168);                    // <=1600 floats
  half_t* Apack  = (half_t*)(ws + 13568);                  // Btot*96 halfs
  half_t* Bpack  = (half_t*)(ws + 13568 + (size_t)Btot * KPAD * sizeof(half_t));

  // 1. graph norm + CSR
  prep_graph_kernel<<<1, 128, 0, stream>>>(edge_index, edge_weight, E,
                                           rowptr, colsrc, val);
  // 2. pack Wfc into WMMA-B fragment order (fp16, K padded to 96)
  {
    int total = 3 * NOUT * 32;
    pack_wfc_kernel<<<(total + 255) / 256, 256, 0, stream>>>(Wfc, Bpack);
  }
  // 3. fused GCN conv1+conv2 -> fp16 A
  gcn_conv_kernel<<<Btot, 128, 0, stream>>>(feature, W1, b1, W2, b2,
                                            rowptr, colsrc, val, EP, Apack);
  // 4. WMMA GEMM + bias -> out[M,6400]
  {
    dim3 grid(NOUT / 64, (Btot / 16 + 3) / 4);
    gemm_wmma_kernel<<<grid, 128, 0, stream>>>(Apack, Bpack, bfc, out, Btot);
  }
}
